// GNNModel_75213467287931
// MI455X (gfx1250) — compile-verified, matrix operand order checked by reading, not run on hardware
//
#include <hip/hip_runtime.h>
#include <hip/hip_bf16.h>

// ---------------- problem constants (match reference) ----------------
#define N_  50000
#define E_  800000
#define G_  64
#define H_  4
#define D_  128
#define E2_ (E_ + N_)   // edges + self loops

typedef _Float16 half_t;
typedef __attribute__((ext_vector_type(16))) _Float16 v16h;
typedef __attribute__((ext_vector_type(8)))  float    v8f;

#define CDIV(a, b) (((a) + (b) - 1) / (b))
#define WN 4   // 16x16 output tiles per wave along N

// ---------------- helpers ----------------
__device__ inline void atomicMaxF(float* a, float v) {
    unsigned int* ua = (unsigned int*)a;
    unsigned int old = *ua;
    while (__uint_as_float(old) < v) {
        unsigned int assumed = old;
        old = atomicCAS(ua, assumed, __float_as_uint(v));
        if (old == assumed) break;
    }
}

// ---------------- B-matrix fragment packing ----------------
// Packs row-major f32 W[K,Nc] into WMMA-B fragment order:
//   [k-tile (K/32)][n-tile (Nc/16)][lane 0..31][16 halves]
// so each lane's v16h fragment is one contiguous 32B load.
// Per ISA 7.12.2 (f16 B 32x16, wave32): lane = hi*16 + (n&15);
//   halves 0..7  -> K = k0 + hi*8 + i
//   halves 8..15 -> K = k0 + 16 + hi*8 + i
__global__ void k_pack_b(const float* __restrict__ w, half_t* __restrict__ bp,
                         int K, int Nc) {
    int i = blockIdx.x * blockDim.x + threadIdx.x;
    if (i >= K * Nc) return;
    int k = i / Nc, n = i % Nc;
    int rem      = k & 31;
    int hi       = (rem >> 3) & 1;
    int half_idx = (rem & 7) + ((rem >> 4) << 3);
    int lane     = hi * 16 + (n & 15);
    int ntile    = Nc >> 4;
    size_t base  = ((size_t)(k >> 5) * ntile + (n >> 4)) * (32 * 16);
    bp[base + lane * 16 + half_idx] = (half_t)w[i];
}

// ---------------- WMMA GEMM ----------------
// C[M,Nc] = act( A[M,K] (f16 row-major) @ Bpacked + bias )
// One wave computes a 16 x (16*WN) output strip; A fragment reused WN times.
__global__ void wmma_gemm_f16(const half_t* __restrict__ A,
                              const half_t* __restrict__ Bpk,
                              const float* __restrict__ bias,
                              float* __restrict__ Cf, half_t* __restrict__ Ch,
                              int M, int Nc, int K, int relu) {
    int wave  = (blockIdx.x * blockDim.x + threadIdx.x) >> 5;
    int ntile = Nc >> 4;
    int ngrp  = ntile / WN;
    int groups = (M >> 4) * ngrp;
    if (wave >= groups) return;            // wave-uniform: EXEC stays all-ones
    int m0 = (wave / ngrp) << 4;
    int ng = (wave % ngrp) * WN;           // first n-tile of this wave's strip
    int lane = threadIdx.x & 31;
    int lr = lane & 15;                    // row (A) / col (B,C) within tile
    int hi = lane >> 4;                    // lane-half selects K sub-block

    const v16h* Bv = (const v16h*)Bpk;
    v8f acc[WN] = {};
    for (int k0 = 0; k0 < K; k0 += 32) {
        v16h a;
        const half_t* Ap = A + (size_t)(m0 + lr) * K + (k0 + hi * 8);
        #pragma unroll
        for (int i = 0; i < 8; ++i) { a[i] = Ap[i]; a[8 + i] = Ap[16 + i]; }
        if (k0 + 32 < K) __builtin_prefetch((const void*)(Ap + 32), 0, 0);
        const v16h* bb = Bv + ((size_t)(k0 >> 5) * ntile + ng) * 32 + lane;
        #pragma unroll
        for (int t = 0; t < WN; ++t) {
            v16h b = bb[t * 32];           // contiguous 32B per lane, coalesced
            acc[t] = __builtin_amdgcn_wmma_f32_16x16x32_f16(
                        /*neg_a=*/false, a, /*neg_b=*/false, b,
                        /*c_mod=*/(short)0, acc[t],
                        /*reuse_a=*/false, /*reuse_b=*/false);
        }
    }

    int mbase = m0 + hi * 8;
    #pragma unroll
    for (int t = 0; t < WN; ++t) {
        int n = ((ng + t) << 4) + lr;
        float bv = bias ? bias[n] : 0.0f;
        #pragma unroll
        for (int r = 0; r < 8; ++r) {
            float v = acc[t][r] + bv;
            if (relu) v = v > 0.f ? v : 0.f;
            size_t idx = (size_t)(mbase + r) * Nc + n;
            if (Cf) Cf[idx] = v;
            if (Ch) Ch[idx] = (half_t)v;
        }
    }
}

// ---------------- small utility kernels ----------------
__global__ void k_fill_emax(float* e_max) {
    int i = blockIdx.x * blockDim.x + threadIdx.x;
    if (i < N_ * H_) e_max[i] = -1e30f;
}

__global__ void k_fill_edges(const int* __restrict__ ei, int* __restrict__ ei_out) {
    int i = blockIdx.x * blockDim.x + threadIdx.x;
    if (i >= E2_) return;
    ei_out[i]       = (i < E_) ? ei[i]      : (i - E_);   // src2
    ei_out[E2_ + i] = (i < E_) ? ei[E_ + i] : (i - E_);   // dst2
}

// GIN1 aggregation: 2 features per edge
__global__ void k_scatter_x(const float* __restrict__ x, const int* __restrict__ ei,
                            float* __restrict__ agg1) {
    int e = blockIdx.x * blockDim.x + threadIdx.x;
    if (e >= E_) return;
    int s = ei[e], d = ei[E_ + e];
    atomicAdd(&agg1[(size_t)d * 2 + 0], x[(size_t)s * 2 + 0]);
    atomicAdd(&agg1[(size_t)d * 2 + 1], x[(size_t)s * 2 + 1]);
}

// GIN1 first layer (K=2, below WMMA granularity): relu((x+agg) @ w1 + b1) -> f16
__global__ void k_gin1_hidden(const float* __restrict__ x, const float* __restrict__ agg1,
                              const float* __restrict__ w1, const float* __restrict__ b1,
                              half_t* __restrict__ hid1h) {
    int i = blockIdx.x * blockDim.x + threadIdx.x;
    if (i >= N_ * 64) return;
    int n = i >> 6, j = i & 63;
    float in0 = x[(size_t)n * 2 + 0] + agg1[(size_t)n * 2 + 0];
    float in1 = x[(size_t)n * 2 + 1] + agg1[(size_t)n * 2 + 1];
    float v = in0 * w1[j] + in1 * w1[64 + j] + b1[j];
    hid1h[i] = (half_t)(v > 0.f ? v : 0.f);
}

// GIN2 aggregation: 64 features per edge
__global__ void k_scatter_h(const float* __restrict__ h1, const int* __restrict__ ei,
                            float* __restrict__ agg2) {
    size_t i = (size_t)blockIdx.x * blockDim.x + threadIdx.x;
    if (i >= (size_t)E_ * 64) return;
    int e = (int)(i >> 6), c = (int)(i & 63);
    int s = ei[e], d = ei[E_ + e];
    atomicAdd(&agg2[(size_t)d * 64 + c], h1[(size_t)s * 64 + c]);
}

__global__ void k_add_to_f16(const float* __restrict__ a, const float* __restrict__ b,
                             half_t* __restrict__ d, int n) {
    int i = blockIdx.x * blockDim.x + threadIdx.x;
    if (i < n) d[i] = (half_t)(a[i] + b[i]);
}

// per-(node,head) attention logits
__global__ void k_att(const float* __restrict__ hw, const float* __restrict__ att_src,
                      const float* __restrict__ att_dst,
                      float* __restrict__ a_src, float* __restrict__ a_dst) {
    int i = blockIdx.x * blockDim.x + threadIdx.x;
    if (i >= N_ * H_) return;
    int n = i >> 2, h = i & 3;
    const float* row = hw + (size_t)n * 512 + h * 128;
    float s = 0.f, t = 0.f;
    for (int d = 0; d < 128; ++d) {
        float v = row[d];
        s += v * att_src[h * 128 + d];
        t += v * att_dst[h * 128 + d];
    }
    a_src[i] = s; a_dst[i] = t;
}

__device__ inline void edge_endpoints(const int* ei, int e, int& s, int& d) {
    if (e < E_) { s = ei[e]; d = ei[E_ + e]; } else { s = e - E_; d = e - E_; }
}

__global__ void k_emax(const int* __restrict__ ei, const float* __restrict__ a_src,
                       const float* __restrict__ a_dst, float* __restrict__ e_max) {
    size_t i = (size_t)blockIdx.x * blockDim.x + threadIdx.x;
    if (i >= (size_t)E2_ * H_) return;
    int e = (int)(i >> 2), h = (int)(i & 3), s, d;
    edge_endpoints(ei, e, s, d);
    float v = a_src[(size_t)s * H_ + h] + a_dst[(size_t)d * H_ + h];
    v = v > 0.f ? v : 0.2f * v;                        // leaky_relu(0.2)
    atomicMaxF(&e_max[(size_t)d * H_ + h], v);
}

__global__ void k_expsum(const int* __restrict__ ei, const float* __restrict__ a_src,
                         const float* __restrict__ a_dst, const float* __restrict__ e_max,
                         float* __restrict__ alpha, float* __restrict__ e_sum) {
    size_t i = (size_t)blockIdx.x * blockDim.x + threadIdx.x;
    if (i >= (size_t)E2_ * H_) return;
    int e = (int)(i >> 2), h = (int)(i & 3), s, d;
    edge_endpoints(ei, e, s, d);
    float v = a_src[(size_t)s * H_ + h] + a_dst[(size_t)d * H_ + h];
    v = v > 0.f ? v : 0.2f * v;
    float ex = __expf(v - e_max[(size_t)d * H_ + h]);
    alpha[(size_t)e * H_ + h] = ex;
    atomicAdd(&e_sum[(size_t)d * H_ + h], ex);
}

__global__ void k_norm(const int* __restrict__ ei, const float* __restrict__ e_sum,
                       float* __restrict__ alpha) {
    size_t i = (size_t)blockIdx.x * blockDim.x + threadIdx.x;
    if (i >= (size_t)E2_ * H_) return;
    int e = (int)(i >> 2), h = (int)(i & 3), s, d;
    edge_endpoints(ei, e, s, d);
    alpha[i] /= e_sum[(size_t)d * H_ + h];
}

__global__ void k_msg(const int* __restrict__ ei, const float* __restrict__ alpha,
                      const float* __restrict__ hw, float* __restrict__ out_nodes) {
    size_t i = (size_t)blockIdx.x * blockDim.x + threadIdx.x;
    if (i >= (size_t)E2_ * 512) return;
    int e = (int)(i >> 9), c = (int)(i & 511), h = c >> 7, s, d;
    edge_endpoints(ei, e, s, d);
    atomicAdd(&out_nodes[(size_t)d * 512 + c],
              alpha[(size_t)e * H_ + h] * hw[(size_t)s * 512 + c]);
}

__global__ void k_counts(const int* __restrict__ batch, float* __restrict__ counts) {
    int n = blockIdx.x * blockDim.x + threadIdx.x;
    if (n < N_) atomicAdd(&counts[batch[n]], 1.0f);
}

__global__ void k_pool(const int* __restrict__ batch, const float* __restrict__ out_nodes,
                       const float* __restrict__ gat_b, float* __restrict__ pooled) {
    size_t i = (size_t)blockIdx.x * blockDim.x + threadIdx.x;
    if (i >= (size_t)N_ * 512) return;
    int n = (int)(i >> 9), c = (int)(i & 511);
    atomicAdd(&pooled[(size_t)batch[n] * 512 + c], out_nodes[i] + gat_b[c]);
}

__global__ void k_pool_div(const float* __restrict__ counts, float* __restrict__ pooled,
                           half_t* __restrict__ pooledh) {
    int i = blockIdx.x * blockDim.x + threadIdx.x;
    if (i >= G_ * 512) return;
    int g = i >> 9;
    float v = pooled[i] / fmaxf(counts[g], 1.0f);
    pooled[i]  = v;
    pooledh[i] = (half_t)v;
}

__global__ void k_out(const float* __restrict__ fc, const float* __restrict__ out_w,
                      const float* __restrict__ out_b, float* __restrict__ y) {
    int i = blockIdx.x * blockDim.x + threadIdx.x;
    if (i >= G_ * 4) return;
    int g = i >> 2, o = i & 3;
    float s = out_b[o];
    for (int k = 0; k < 512; ++k) s += fc[(size_t)g * 512 + k] * out_w[k * 4 + o];
    y[i] = s;
}

// ---------------- launch ----------------
extern "C" void kernel_launch(void* const* d_in, const int* in_sizes, int n_in,
                              void* d_out, int out_size, void* d_ws, size_t ws_size,
                              hipStream_t stream) {
    const float* x       = (const float*)d_in[0];
    const int*   ei      = (const int*)  d_in[1];
    const int*   batch   = (const int*)  d_in[2];
    const float* g1w1    = (const float*)d_in[3];
    const float* g1b1    = (const float*)d_in[4];
    const float* g1w2    = (const float*)d_in[5];
    const float* g1b2    = (const float*)d_in[6];
    const float* g2w1    = (const float*)d_in[7];
    const float* g2b1    = (const float*)d_in[8];
    const float* g2w2    = (const float*)d_in[9];
    const float* g2b2    = (const float*)d_in[10];
    const float* gat_w   = (const float*)d_in[11];
    const float* att_src = (const float*)d_in[12];
    const float* att_dst = (const float*)d_in[13];
    const float* gat_b   = (const float*)d_in[14];
    const float* fc_w    = (const float*)d_in[15];
    const float* fc_b    = (const float*)d_in[16];
    const float* out_w   = (const float*)d_in[17];
    const float* out_b   = (const float*)d_in[18];

    // ---- d_out layout: y[G*4] | edge_index_sl[2*E2] (int bits) | alpha[E2*H]
    float* y      = (float*)d_out;
    int*   ei_out = (int*)((float*)d_out + (size_t)G_ * 4);
    float* alpha  = (float*)d_out + (size_t)G_ * 4 + (size_t)2 * E2_;

    // ---- carve workspace (256B aligned)
    char* p = (char*)d_ws;
    auto carve = [&](size_t bytes) { void* r = (void*)p; p += (bytes + 255) & ~(size_t)255; return r; };
    float*  agg1      = (float*) carve((size_t)N_ * 2   * 4);
    half_t* hid1h     = (half_t*)carve((size_t)N_ * 64  * 2);
    float*  h1        = (float*) carve((size_t)N_ * 64  * 4);
    float*  agg2      = (float*) carve((size_t)N_ * 64  * 4);
    half_t* t2h       = (half_t*)carve((size_t)N_ * 64  * 2);
    half_t* hid2h     = (half_t*)carve((size_t)N_ * 64  * 2);
    half_t* h2h       = (half_t*)carve((size_t)N_ * 128 * 2);
    float*  hw        = (float*) carve((size_t)N_ * 512 * 4);
    float*  a_src     = (float*) carve((size_t)N_ * H_  * 4);
    float*  a_dst     = (float*) carve((size_t)N_ * H_  * 4);
    float*  e_max     = (float*) carve((size_t)N_ * H_  * 4);
    float*  e_sum     = (float*) carve((size_t)N_ * H_  * 4);
    float*  out_nodes = (float*) carve((size_t)N_ * 512 * 4);
    float*  pooled    = (float*) carve((size_t)G_ * 512 * 4);
    float*  counts    = (float*) carve((size_t)G_ * 4);
    half_t* pooledh   = (half_t*)carve((size_t)G_ * 512 * 2);
    float*  fc_out    = (float*) carve((size_t)G_ * 512 * 4);
    half_t* g1w2p     = (half_t*)carve((size_t)64 * 64  * 2);
    half_t* g2w1p     = (half_t*)carve((size_t)64 * 64  * 2);
    half_t* g2w2p     = (half_t*)carve((size_t)64 * 128 * 2);
    half_t* gat_wp    = (half_t*)carve((size_t)128 * 512 * 2);
    half_t* fc_wp     = (half_t*)carve((size_t)512 * 512 * 2);

    const int B = 256;

    // ---- zero accumulators (capture-safe)
    hipMemsetAsync(agg1,      0, (size_t)N_ * 2   * 4, stream);
    hipMemsetAsync(agg2,      0, (size_t)N_ * 64  * 4, stream);
    hipMemsetAsync(e_sum,     0, (size_t)N_ * H_  * 4, stream);
    hipMemsetAsync(out_nodes, 0, (size_t)N_ * 512 * 4, stream);
    hipMemsetAsync(pooled,    0, (size_t)G_ * 512 * 4, stream);
    hipMemsetAsync(counts,    0, (size_t)G_ * 4,       stream);

    k_fill_emax <<<CDIV(N_ * H_, B), B, 0, stream>>>(e_max);
    k_fill_edges<<<CDIV(E2_, B), B, 0, stream>>>(ei, ei_out);

    // ---- weight conversion + fragment packing (f32 -> f16 WMMA-B layout)
    k_pack_b<<<CDIV(64 * 64, B),   B, 0, stream>>>(g1w2,  g1w2p, 64, 64);
    k_pack_b<<<CDIV(64 * 64, B),   B, 0, stream>>>(g2w1,  g2w1p, 64, 64);
    k_pack_b<<<CDIV(64 * 128, B),  B, 0, stream>>>(g2w2,  g2w2p, 64, 128);
    k_pack_b<<<CDIV(128 * 512, B), B, 0, stream>>>(gat_w, gat_wp, 128, 512);
    k_pack_b<<<CDIV(512 * 512, B), B, 0, stream>>>(fc_w,  fc_wp, 512, 512);

    // ---- GIN1
    k_scatter_x  <<<CDIV(E_, B), B, 0, stream>>>(x, ei, agg1);
    k_gin1_hidden<<<CDIV(N_ * 64, B), B, 0, stream>>>(x, agg1, g1w1, g1b1, hid1h);
    {   // h1 = relu(hid1 @ g1w2 + g1b2)   [N,64]
        int groups = (N_ / 16) * (64 / 16 / WN);
        wmma_gemm_f16<<<CDIV(groups * 32, B), B, 0, stream>>>(
            hid1h, g1w2p, g1b2, h1, nullptr, N_, 64, 64, 1);
    }

    // ---- GIN2
    k_scatter_h <<<CDIV((size_t)E_ * 64, B), B, 0, stream>>>(h1, ei, agg2);
    k_add_to_f16<<<CDIV(N_ * 64, B), B, 0, stream>>>(h1, agg2, t2h, N_ * 64);
    {   // hid2 = relu(t2 @ g2w1 + g2b1)   [N,64] -> f16
        int groups = (N_ / 16) * (64 / 16 / WN);
        wmma_gemm_f16<<<CDIV(groups * 32, B), B, 0, stream>>>(
            t2h, g2w1p, g2b1, nullptr, hid2h, N_, 64, 64, 1);
    }
    {   // h2 = relu(hid2 @ g2w2 + g2b2)   [N,128] -> f16
        int groups = (N_ / 16) * (128 / 16 / WN);
        wmma_gemm_f16<<<CDIV(groups * 32, B), B, 0, stream>>>(
            hid2h, g2w2p, g2b2, nullptr, h2h, N_, 128, 64, 1);
    }

    // ---- GAT projection: hw = h2 @ gat_w   [N,512]
    {
        int groups = (N_ / 16) * (512 / 16 / WN);
        wmma_gemm_f16<<<CDIV((size_t)groups * 32, B), B, 0, stream>>>(
            h2h, gat_wp, nullptr, hw, nullptr, N_, 512, 128, 0);
    }

    // ---- attention + segment softmax
    k_att   <<<CDIV(N_ * H_, B), B, 0, stream>>>(hw, att_src, att_dst, a_src, a_dst);
    k_emax  <<<CDIV((size_t)E2_ * H_, B), B, 0, stream>>>(ei, a_src, a_dst, e_max);
    k_expsum<<<CDIV((size_t)E2_ * H_, B), B, 0, stream>>>(ei, a_src, a_dst, e_max, alpha, e_sum);
    k_norm  <<<CDIV((size_t)E2_ * H_, B), B, 0, stream>>>(ei, e_sum, alpha);

    // ---- message passing + mean pool
    k_msg   <<<CDIV((size_t)E2_ * 512, B), B, 0, stream>>>(ei, alpha, hw, out_nodes);
    k_counts<<<CDIV(N_, B), B, 0, stream>>>(batch, counts);
    k_pool  <<<CDIV((size_t)N_ * 512, B), B, 0, stream>>>(batch, out_nodes, gat_b, pooled);
    k_pool_div<<<CDIV(G_ * 512, B), B, 0, stream>>>(counts, pooled, pooledh);

    // ---- head
    {   // fc_out = relu(pooled @ fc_w + fc_b)   [G,512]
        int groups = (G_ / 16) * (512 / 16 / WN);
        wmma_gemm_f16<<<CDIV(groups * 32, B), B, 0, stream>>>(
            pooledh, fc_wp, fc_b, fc_out, nullptr, G_, 512, 512, 1);
    }
    k_out<<<CDIV(G_ * 4, B), B, 0, stream>>>(fc_out, out_w, out_b, y);

    (void)in_sizes; (void)n_in; (void)out_size; (void)ws_size;
}
